// SpineSegmentationNet_88029649699281
// MI455X (gfx1250) — compile-verified, matrix operand order checked by reading, not run on hardware
//
#include <hip/hip_runtime.h>
#include <hip/hip_bf16.h>
#include <math.h>

// ---------------- problem constants ----------------
constexpr int BGR  = 2;      // graphs
constexpr int NPER = 8192;   // points per graph
constexpr int NS1  = 4096;   // SA1 samples per graph
constexpr int NS2  = 1024;   // SA2 samples per graph
constexpr int KNB  = 64;     // max neighbors
constexpr int Q1   = BGR * NS1;        // 8192 queries SA1
constexpr int Q2   = BGR * NS2;        // 2048 queries SA2
constexpr long M1  = (long)Q1 * KNB;   // 524288 edge rows SA1
constexpr long M2  = (long)Q2 * KNB;   // 131072 edge rows SA2

typedef __attribute__((ext_vector_type(16))) _Float16 v16h_t;
typedef __attribute__((ext_vector_type(8)))  _Float16 v8h_t;
typedef __attribute__((ext_vector_type(8)))  float    v8f_t;

// -------- WMMA fragment loader --------
// Data stored row-major [16 rows x Kpad] f16.  Per the CDNA5 16-bit A/B layout:
//  lanes 0-15 : row = lane,    elements 0-7 = K(kb+0..7),  8-15 = K(kb+16..23)
//  lanes 16-31: row = lane-16, elements 0-7 = K(kb+8..15), 8-15 = K(kb+24..31)
// -> two contiguous 16-byte loads per lane.
__device__ __forceinline__ v16h_t load_frag16(const _Float16* __restrict__ base,
                                              int ld, int kb, int lane) {
  int r  = lane & 15;
  int ko = (lane < 16) ? 0 : 8;
  const _Float16* p = base + (long)r * ld + kb + ko;
  v8h_t lo = *(const v8h_t*)(p);
  v8h_t hi = *(const v8h_t*)(p + 16);
  return __builtin_shufflevector(lo, hi, 0,1,2,3,4,5,6,7,8,9,10,11,12,13,14,15);
}

// ---------------- FPS: one block per graph ----------------
__global__ void fps_kernel(const float* __restrict__ pos, int n, int S,
                           int* __restrict__ idx, float* __restrict__ dist) {
  const int b = blockIdx.x;
  const float* P = pos + (long)b * n * 3;
  int*   I = idx  + (long)b * S;
  float* D = dist + (long)b * n;
  __shared__ float sM[256];
  __shared__ int   sI[256];
  __shared__ int   sLast;
  const int tid = threadIdx.x;
  for (int i = tid; i < n; i += 256) D[i] = 3.4e38f;
  if (tid == 0) { sLast = 0; I[0] = 0; }
  __syncthreads();
  for (int it = 1; it < S; ++it) {
    const int last = sLast;
    const float lx = P[last*3+0], ly = P[last*3+1], lz = P[last*3+2];
    float bm = -1.0f; int bi = 0;
    for (int i = tid; i < n; i += 256) {
      float dx = P[i*3+0]-lx, dy = P[i*3+1]-ly, dz = P[i*3+2]-lz;
      float d = fminf(D[i], dx*dx + dy*dy + dz*dz);
      D[i] = d;
      if (d > bm) { bm = d; bi = i; }
    }
    sM[tid] = bm; sI[tid] = bi;
    __syncthreads();
    for (int s = 128; s > 0; s >>= 1) {
      if (tid < s) {
        if (sM[tid+s] > sM[tid] ||
            (sM[tid+s] == sM[tid] && sI[tid+s] < sI[tid])) {
          sM[tid] = sM[tid+s]; sI[tid] = sI[tid+s];
        }
      }
      __syncthreads();
    }
    if (tid == 0) { I[it] = sI[0]; sLast = sI[0]; }
    __syncthreads();
  }
}

// ---------------- gather sampled positions ----------------
__global__ void gather3_kernel(const float* __restrict__ pos, const int* __restrict__ idx,
                               float* __restrict__ q, int n, int S) {
  int t = blockIdx.x * blockDim.x + threadIdx.x;        // over B*S
  if (t >= BGR * S) return;
  int b = t / S;
  int j = idx[t];
  const float* p = pos + ((long)b * n + j) * 3;
  q[t*3+0] = p[0]; q[t*3+1] = p[1]; q[t*3+2] = p[2];
}

// ---------------- radius neighbors: one wave32 per query ----------------
__global__ void radius_kernel(const float* __restrict__ qpos, const float* __restrict__ ppos,
                              int Sq, int n, float r2,
                              int* __restrict__ nbr, int* __restrict__ cnt) {
  const int lane = threadIdx.x & 31;
  const int qi = blockIdx.x * 8 + (threadIdx.x >> 5);   // grid sized exactly
  const int b = qi / Sq;
  const float* P = ppos + (long)b * n * 3;
  const float qx = qpos[qi*3+0], qy = qpos[qi*3+1], qz = qpos[qi*3+2];
  int* NB = nbr + (long)qi * KNB;
  int count = 0;
  for (int base = 0; base < n; base += 32) {
    const int i = base + lane;                          // n multiple of 32
    float dx = P[i*3+0]-qx, dy = P[i*3+1]-qy, dz = P[i*3+2]-qz;
    bool valid = (dx*dx + dy*dy + dz*dz) <= r2;
    unsigned mask = (unsigned)__ballot(valid);
    int slot = count + __popc(mask & ((1u << lane) - 1u));
    if (valid && slot < KNB) NB[slot] = i;
    count += __popc(mask);
    if (count >= KNB) break;
  }
  if (count > KNB) count = KNB;
  if (lane == 0) cnt[qi] = count;
  for (int k = count + lane; k < KNB; k += 32) NB[k] = 0;  // safe gathers
}

// ---------------- SA1 edge features: rel pos, padded to 32 ----------------
__global__ void feat1_kernel(const float* __restrict__ pos, const float* __restrict__ q1,
                             const int* __restrict__ nbr, _Float16* __restrict__ feat,
                             int n, int Sq) {
  long e = (long)blockIdx.x * blockDim.x + threadIdx.x;  // M1 threads
  long qi = e >> 6;
  int  b  = (int)(qi / Sq);
  int  j  = nbr[e];
  const float* p = pos + ((long)b * n + j) * 3;
  float rx = p[0] - q1[qi*3+0];
  float ry = p[1] - q1[qi*3+1];
  float rz = p[2] - q1[qi*3+2];
  _Float16* F = feat + e * 32;
  F[0] = (_Float16)rx; F[1] = (_Float16)ry; F[2] = (_Float16)rz;
  #pragma unroll
  for (int c = 3; c < 32; ++c) F[c] = (_Float16)0.f;
}

// ---------------- SA2 edge features: [x1(128) | rel(3)], padded to 160 ----------------
__global__ void feat2_kernel(const float* __restrict__ ppos, const float* __restrict__ q2,
                             const int* __restrict__ nbr, const _Float16* __restrict__ x1,
                             _Float16* __restrict__ feat, int n, int Sq) {
  long e = (long)blockIdx.x * blockDim.x + threadIdx.x;  // M2 threads
  long qi = e >> 6;
  int  b  = (int)(qi / Sq);
  int  j  = nbr[e];
  const v8h_t* X = (const v8h_t*)(x1 + ((long)b * n + j) * 128);
  v8h_t* F = (v8h_t*)(feat + e * 160);
  #pragma unroll
  for (int v = 0; v < 16; ++v) F[v] = X[v];             // 128 f16 feature copy
  const float* p = ppos + ((long)b * n + j) * 3;
  _Float16* Ft = feat + e * 160;
  Ft[128] = (_Float16)(p[0] - q2[qi*3+0]);
  Ft[129] = (_Float16)(p[1] - q2[qi*3+1]);
  Ft[130] = (_Float16)(p[2] - q2[qi*3+2]);
  #pragma unroll
  for (int c = 131; c < 160; ++c) Ft[c] = (_Float16)0.f;
}

// ---------------- weight prep: W[Kin,N] f32 -> Wt[N,Kpad] f16 (transpose+pad) ----------------
__global__ void wprep_kernel(const float* __restrict__ W, _Float16* __restrict__ Wt,
                             int Kin, int N, int Kpad) {
  int t = blockIdx.x * blockDim.x + threadIdx.x;
  if (t >= N * Kpad) return;
  int n = t / Kpad, k = t - n * Kpad;
  Wt[t] = (k < Kin) ? (_Float16)W[(long)k * N + n] : (_Float16)0.f;
}

// ---------------- generic WMMA GEMM + bias + relu, f16 in / f16 out ----------------
// A [M,Kpad] row-major f16, Wt [N,Kpad] row-major f16 (i.e. W^T).
// block = 256 (8 waves); each wave computes 16 rows x 64 cols (4 accumulators,
// one A fragment shared across 4 B fragments per k-step -> 4x A reuse).
// Block covers 128 rows x 64 cols; grid = (M/128, N/64).
__global__ void gemm_relu_wmma(const _Float16* __restrict__ A, const _Float16* __restrict__ Wt,
                               const float* __restrict__ bias, _Float16* __restrict__ out,
                               int Kpad, int N) {
  const int wave = threadIdx.x >> 5;
  const int lane = threadIdx.x & 31;
  const long row0 = ((long)blockIdx.x * 8 + wave) * 16;
  const int  col0 = blockIdx.y * 64;
  const _Float16* Ab = A + row0 * Kpad;
  const _Float16* Wb = Wt + (long)col0 * Kpad;
  v8f_t acc0 = {}, acc1 = {}, acc2 = {}, acc3 = {};
  for (int kb = 0; kb < Kpad; kb += 32) {
    v16h_t a  = load_frag16(Ab, Kpad, kb, lane);
    v16h_t b0 = load_frag16(Wb + 0L  * 16 * Kpad, Kpad, kb, lane);
    v16h_t b1 = load_frag16(Wb + 1L  * 16 * Kpad, Kpad, kb, lane);
    v16h_t b2 = load_frag16(Wb + 2L  * 16 * Kpad, Kpad, kb, lane);
    v16h_t b3 = load_frag16(Wb + 3L  * 16 * Kpad, Kpad, kb, lane);
    acc0 = __builtin_amdgcn_wmma_f32_16x16x32_f16(false, a, false, b0, (short)0, acc0, false, false);
    acc1 = __builtin_amdgcn_wmma_f32_16x16x32_f16(false, a, false, b1, (short)0, acc1, false, false);
    acc2 = __builtin_amdgcn_wmma_f32_16x16x32_f16(false, a, false, b2, (short)0, acc2, false, false);
    acc3 = __builtin_amdgcn_wmma_f32_16x16x32_f16(false, a, false, b3, (short)0, acc3, false, false);
  }
  const long rbase = row0 + ((lane < 16) ? 0 : 8);
  const int  cl    = lane & 15;
  v8f_t accs[4] = {acc0, acc1, acc2, acc3};
  #pragma unroll
  for (int t = 0; t < 4; ++t) {
    const int col = col0 + t * 16 + cl;
    const float bv = bias[col];
    #pragma unroll
    for (int i = 0; i < 8; ++i) {
      float v = fmaxf(accs[t][i] + bv, 0.f);
      out[(rbase + i) * N + col] = (_Float16)v;
    }
  }
}

// ---------------- fused final-layer GEMM + bias + mask + max over K=64 ----------------
// block = 128 (4 waves) covers one query's 64 edge rows; each wave handles 64 cols
// (4 accumulators sharing one A fragment).  grid = (Q, N/64).  Writes x[q][col] f16.
__global__ void gemm_maxk_wmma(const _Float16* __restrict__ A, const _Float16* __restrict__ Wt,
                               const float* __restrict__ bias, const int* __restrict__ cnt,
                               _Float16* __restrict__ xout, int Kpad, int N) {
  __shared__ float part[4][4][32];                      // [wave][colgroup][lane]
  const int wave = threadIdx.x >> 5;
  const int lane = threadIdx.x & 31;
  const long q   = blockIdx.x;
  const long row0 = q * KNB + wave * 16;
  const int  col0 = blockIdx.y * 64;
  const _Float16* Ab = A + row0 * Kpad;
  const _Float16* Wb = Wt + (long)col0 * Kpad;
  v8f_t acc0 = {}, acc1 = {}, acc2 = {}, acc3 = {};
  for (int kb = 0; kb < Kpad; kb += 32) {
    v16h_t a  = load_frag16(Ab, Kpad, kb, lane);
    v16h_t b0 = load_frag16(Wb + 0L  * 16 * Kpad, Kpad, kb, lane);
    v16h_t b1 = load_frag16(Wb + 1L  * 16 * Kpad, Kpad, kb, lane);
    v16h_t b2 = load_frag16(Wb + 2L  * 16 * Kpad, Kpad, kb, lane);
    v16h_t b3 = load_frag16(Wb + 3L  * 16 * Kpad, Kpad, kb, lane);
    acc0 = __builtin_amdgcn_wmma_f32_16x16x32_f16(false, a, false, b0, (short)0, acc0, false, false);
    acc1 = __builtin_amdgcn_wmma_f32_16x16x32_f16(false, a, false, b1, (short)0, acc1, false, false);
    acc2 = __builtin_amdgcn_wmma_f32_16x16x32_f16(false, a, false, b2, (short)0, acc2, false, false);
    acc3 = __builtin_amdgcn_wmma_f32_16x16x32_f16(false, a, false, b3, (short)0, acc3, false, false);
  }
  const int c    = cnt[q];
  const int rloc = wave * 16 + ((lane < 16) ? 0 : 8);   // k index within query
  const int cl   = lane & 15;
  v8f_t accs[4] = {acc0, acc1, acc2, acc3};
  #pragma unroll
  for (int t = 0; t < 4; ++t) {
    const float bv = bias[col0 + t * 16 + cl];
    float m = -INFINITY;
    #pragma unroll
    for (int i = 0; i < 8; ++i) {
      float v = accs[t][i] + bv;
      if (rloc + i < c) m = fmaxf(m, v);                // mask invalid neighbors
    }
    part[wave][t][lane] = m;
  }
  __syncthreads();
  if (threadIdx.x < 64) {
    const int t = threadIdx.x >> 4;
    const int cc = threadIdx.x & 15;
    float mm = -INFINITY;
    #pragma unroll
    for (int w = 0; w < 4; ++w)
      mm = fmaxf(mm, fmaxf(part[w][t][cc], part[w][t][cc + 16]));
    xout[q * N + col0 + t * 16 + cc] = (_Float16)mm;
  }
}

// ---------------- head: 128 -> 1 + sigmoid ----------------
__global__ void lin3_sigmoid_kernel(const _Float16* __restrict__ H, const float* __restrict__ W,
                                    const float* __restrict__ b, float* __restrict__ out) {
  int r = blockIdx.x * blockDim.x + threadIdx.x;        // Q2 threads
  if (r >= Q2) return;
  float acc = b[0];
  const _Float16* h = H + (long)r * 128;
  #pragma unroll 8
  for (int c = 0; c < 128; ++c) acc += (float)h[c] * W[c];
  out[r] = 1.f / (1.f + expf(-acc));
}

// ---------------- host launcher ----------------
extern "C" void kernel_launch(void* const* d_in, const int* in_sizes, int n_in,
                              void* d_out, int out_size, void* d_ws, size_t ws_size,
                              hipStream_t stream) {
  const float* pos     = (const float*)d_in[0];
  const float* sa1_W1  = (const float*)d_in[2];  const float* sa1_b1 = (const float*)d_in[3];
  const float* sa1_W2  = (const float*)d_in[4];  const float* sa1_b2 = (const float*)d_in[5];
  const float* sa1_W3  = (const float*)d_in[6];  const float* sa1_b3 = (const float*)d_in[7];
  const float* sa2_W1  = (const float*)d_in[8];  const float* sa2_b1 = (const float*)d_in[9];
  const float* sa2_W2  = (const float*)d_in[10]; const float* sa2_b2 = (const float*)d_in[11];
  const float* sa2_W3  = (const float*)d_in[12]; const float* sa2_b3 = (const float*)d_in[13];
  const float* lin1_W  = (const float*)d_in[14]; const float* lin1_b = (const float*)d_in[15];
  const float* lin2_W  = (const float*)d_in[16]; const float* lin2_b = (const float*)d_in[17];
  const float* lin3_W  = (const float*)d_in[18]; const float* lin3_b = (const float*)d_in[19];
  float* outp = (float*)d_out;

  // workspace carve-out
  char* base = (char*)d_ws;
  size_t off = 0;
  auto alloc = [&](size_t bytes) { size_t r = off; off = (off + bytes + 255) & ~(size_t)255; return r; };
  size_t o_dist  = alloc((size_t)BGR * NPER * 4);
  size_t o_idx1  = alloc((size_t)Q1 * 4);
  size_t o_q1    = alloc((size_t)Q1 * 3 * 4);
  size_t o_nbr1  = alloc((size_t)M1 * 4);
  size_t o_cnt1  = alloc((size_t)Q1 * 4);
  size_t o_idx2  = alloc((size_t)Q2 * 4);
  size_t o_q2    = alloc((size_t)Q2 * 3 * 4);
  size_t o_nbr2  = alloc((size_t)M2 * 4);
  size_t o_cnt2  = alloc((size_t)Q2 * 4);
  size_t o_x1    = alloc((size_t)Q1 * 128 * 2);
  size_t o_x2    = alloc((size_t)Q2 * 256 * 2);
  size_t o_w11   = alloc(64u  * 32  * 2);
  size_t o_w12   = alloc(64u  * 64  * 2);
  size_t o_w13   = alloc(128u * 64  * 2);
  size_t o_w21   = alloc(128u * 160 * 2);
  size_t o_w22   = alloc(128u * 128 * 2);
  size_t o_w23   = alloc(256u * 128 * 2);
  size_t o_wl1   = alloc(256u * 256 * 2);
  size_t o_wl2   = alloc(128u * 256 * 2);
  size_t o_featA = alloc((size_t)M2 * 160 * 2);   // >= M1*32*2 as well
  size_t o_hB    = alloc((size_t)M1 * 64 * 2);    // >= M2*128*2, >= 2048*256*2
  size_t o_hC    = alloc((size_t)M1 * 64 * 2);

  float*     dist  = (float*)(base + o_dist);
  int*       idx1  = (int*)(base + o_idx1);
  float*     q1    = (float*)(base + o_q1);
  int*       nbr1  = (int*)(base + o_nbr1);
  int*       cnt1  = (int*)(base + o_cnt1);
  int*       idx2  = (int*)(base + o_idx2);
  float*     q2    = (float*)(base + o_q2);
  int*       nbr2  = (int*)(base + o_nbr2);
  int*       cnt2  = (int*)(base + o_cnt2);
  _Float16*  x1    = (_Float16*)(base + o_x1);
  _Float16*  x2    = (_Float16*)(base + o_x2);
  _Float16*  w11   = (_Float16*)(base + o_w11);
  _Float16*  w12   = (_Float16*)(base + o_w12);
  _Float16*  w13   = (_Float16*)(base + o_w13);
  _Float16*  w21   = (_Float16*)(base + o_w21);
  _Float16*  w22   = (_Float16*)(base + o_w22);
  _Float16*  w23   = (_Float16*)(base + o_w23);
  _Float16*  wl1   = (_Float16*)(base + o_wl1);
  _Float16*  wl2   = (_Float16*)(base + o_wl2);
  _Float16*  featA = (_Float16*)(base + o_featA);
  _Float16*  hB    = (_Float16*)(base + o_hB);
  _Float16*  hC    = (_Float16*)(base + o_hC);

  auto gup = [](int total) { return (total + 255) / 256; };

  // ---- weight transposes (f32 -> f16 [N,Kpad]) ----
  wprep_kernel<<<gup(64*32),   256, 0, stream>>>(sa1_W1, w11,   3,  64,  32);
  wprep_kernel<<<gup(64*64),   256, 0, stream>>>(sa1_W2, w12,  64,  64,  64);
  wprep_kernel<<<gup(128*64),  256, 0, stream>>>(sa1_W3, w13,  64, 128,  64);
  wprep_kernel<<<gup(128*160), 256, 0, stream>>>(sa2_W1, w21, 131, 128, 160);
  wprep_kernel<<<gup(128*128), 256, 0, stream>>>(sa2_W2, w22, 128, 128, 128);
  wprep_kernel<<<gup(256*128), 256, 0, stream>>>(sa2_W3, w23, 128, 256, 128);
  wprep_kernel<<<gup(256*256), 256, 0, stream>>>(lin1_W, wl1, 256, 256, 256);
  wprep_kernel<<<gup(128*256), 256, 0, stream>>>(lin2_W, wl2, 256, 128, 256);

  // ---- SA1 ----
  fps_kernel<<<BGR, 256, 0, stream>>>(pos, NPER, NS1, idx1, dist);
  gather3_kernel<<<gup(Q1), 256, 0, stream>>>(pos, idx1, q1, NPER, NS1);
  radius_kernel<<<Q1 / 8, 256, 0, stream>>>(q1, pos, NS1, NPER, 0.2f * 0.2f, nbr1, cnt1);
  feat1_kernel<<<(int)(M1 / 256), 256, 0, stream>>>(pos, q1, nbr1, featA, NPER, NS1);
  gemm_relu_wmma<<<dim3((int)(M1 / 128), 64 / 64), 256, 0, stream>>>(featA, w11, sa1_b1, hB, 32, 64);
  gemm_relu_wmma<<<dim3((int)(M1 / 128), 64 / 64), 256, 0, stream>>>(hB, w12, sa1_b2, hC, 64, 64);
  gemm_maxk_wmma<<<dim3(Q1, 128 / 64), 128, 0, stream>>>(hC, w13, sa1_b3, cnt1, x1, 64, 128);

  // ---- SA2 ----
  fps_kernel<<<BGR, 256, 0, stream>>>(q1, NS1, NS2, idx2, dist);
  gather3_kernel<<<gup(Q2), 256, 0, stream>>>(q1, idx2, q2, NS1, NS2);
  radius_kernel<<<Q2 / 8, 256, 0, stream>>>(q2, q1, NS2, NS1, 0.4f * 0.4f, nbr2, cnt2);
  feat2_kernel<<<(int)(M2 / 256), 256, 0, stream>>>(q1, q2, nbr2, x1, featA, NS1, NS2);
  gemm_relu_wmma<<<dim3((int)(M2 / 128), 128 / 64), 256, 0, stream>>>(featA, w21, sa2_b1, hB, 160, 128);
  gemm_relu_wmma<<<dim3((int)(M2 / 128), 128 / 64), 256, 0, stream>>>(hB, w22, sa2_b2, hC, 128, 128);
  gemm_maxk_wmma<<<dim3(Q2, 256 / 64), 128, 0, stream>>>(hC, w23, sa2_b3, cnt2, x2, 128, 256);

  // ---- head MLP ----
  gemm_relu_wmma<<<dim3(Q2 / 128, 256 / 64), 256, 0, stream>>>(x2, wl1, lin1_b, hB, 256, 256);
  gemm_relu_wmma<<<dim3(Q2 / 128, 128 / 64), 256, 0, stream>>>(hB, wl2, lin2_b, hC, 256, 128);
  lin3_sigmoid_kernel<<<gup(Q2), 256, 0, stream>>>(hC, lin3_W, lin3_b, outp);
}
